// MSBGHearingModel_36438502539758
// MI455X (gfx1250) — matrix operand order repaired and checked by reading, not compiled, not run on hardware
//
#include <hip/hip_runtime.h>
#include <hip/hip_bf16.h>
#include <math.h>

// ---------------- problem constants (match reference setup_inputs) ----------
constexpr int   T_LEN = 220500;          // samples per batch
constexpr int   BATCH = 2;
constexpr int   HOPC  = 64;
constexpr int   WINC  = 256;             // smear window length
constexpr int   NFC   = T_LEN / HOPC + 1;          // 3446 frames per batch
constexpr int   RROWS = BATCH * NFC;               // 6892 frame rows
constexpr int   RP    = 6896;                      // padded to multiple of 16
constexpr int   NB    = 256;                       // spectral bins used in matmuls
constexpr int   KS    = 1025;                      // gammatone / lpf taps
constexpr int   KP    = 1056;                      // taps padded to 33*32
constexpr int   NKC   = KP / 32;                   // 33 K-chunks
constexpr int   CH    = 28;
constexpr int   CHP   = 32;                        // channels padded to 2*16
constexpr int   HALO  = 1056;
constexpr int   UPLEN = ((T_LEN + 2 * HALO + 31) / 32) * 32;  // 222624, 32-elem aligned rows
constexpr int   TT16  = (T_LEN + 15) / 16;         // 16-sample time tiles per batch
constexpr int   T2048 = (T_LEN + 2047) / 2048;     // 108 LPF mega-tiles per (b,c)
constexpr int   BTOE_CH = NKC * 512;               // 16896 halves per channel Toeplitz
constexpr float EPSF  = 1e-8f;
constexpr float PIF   = 3.14159265358979323846f;

typedef __attribute__((ext_vector_type(16))) _Float16 v16h;
typedef __attribute__((ext_vector_type(8)))  _Float16 v8h;
typedef __attribute__((ext_vector_type(8)))  float    v8f;

#define WMMA_F16(a, b, c) \
  __builtin_amdgcn_wmma_f32_16x16x32_f16(false, (a), false, (b), (short)0, (c), false, false)

__device__ inline v16h cat8(v8h a, v8h b) {
  return __builtin_shufflevector(a, b, 0,1,2,3,4,5,6,7,8,9,10,11,12,13,14,15);
}

// ---------------- WMMA tile loaders (contiguous k order, same for A and B) -----
// A (16x32 f16): lane L -> M = L%16; lanes 0-15 hold K {0..15},
// lanes 16-31 hold K {16..31}: one contiguous 16-half run -> 2x b128.
// Requires 16B alignment: A base 16B-aligned, lda multiple of 8, kbase mult 32.
__device__ inline v16h load_A_tile(const _Float16* __restrict__ A, int lda,
                                   int mbase, int kbase, int lane) {
  const int m   = mbase + (lane & 15);
  const int hi16 = ((lane >> 4) & 1) * 16;
  const _Float16* p = A + (long)m * lda + kbase + hi16;
  v8h lo = *(const v8h*)p;
  v8h hh = *(const v8h*)(p + 8);
  return cat8(lo, hh);
}

// Swizzled-B storage: element (k, n) of a KxN matrix lives at
//   ((((k>>5)*2 + ((k>>4)&1)) * N + n) << 4) + (k & 15)
// so the 16 halves lane (n, hi) needs for chunk kt are contiguous (32B aligned).
__host__ __device__ inline int sw_idx(int k, int n, int N) {
  return ((((k >> 5) * 2 + ((k >> 4) & 1)) * N + n) << 4) + (k & 15);
}

__device__ inline v16h load_B_sw(const _Float16* __restrict__ Bsw, int N,
                                 int kt, int nbase, int lane) {
  const int n  = nbase + (lane & 15);
  const int hi = (lane >> 4) & 1;
  const _Float16* p = Bsw + ((long)((kt * 2 + hi) * N + n) << 4);
  v8h lo = *(const v8h*)p;
  v8h hh = *(const v8h*)(p + 8);
  return cat8(lo, hh);
}

// ---------------- generic helpers ------------------------------------------
__global__ void zero_half(_Float16* __restrict__ p, long n) {
  long i = (long)blockIdx.x * blockDim.x + threadIdx.x;
  if (i < n) p[i] = (_Float16)0.0f;
}

// 'same' cross-correlation (XLA conv has no kernel flip)
__global__ void conv1d_same(const float* __restrict__ in, const float* __restrict__ w,
                            float* __restrict__ out, int n, int taps) {
  long i = (long)blockIdx.x * blockDim.x + threadIdx.x;
  long total = (long)BATCH * n;
  if (i >= total) return;
  int b = (int)(i / n), t = (int)(i - (long)b * n);
  int pad = taps / 2;
  const float* x = in + (long)b * n;
  float acc = 0.f;
  for (int k = 0; k < taps; ++k) {
    int s = t + k - pad;
    if (s >= 0 && s < n) acc = fmaf(x[s], w[k], acc);
  }
  out[i] = acc;
}

// ---------------- STFT stage -------------------------------------------------
__global__ void build_frames(const float* __restrict__ y1,
                             const float* __restrict__ win,
                             _Float16* __restrict__ frames) {
  long idx = (long)blockIdx.x * blockDim.x + threadIdx.x;
  if (idx >= (long)RP * WINC) return;
  int r = (int)(idx / WINC), t = (int)(idx - (long)r * WINC);
  float v = 0.f;
  if (r < RROWS) {
    int b = r / NFC, f = r - b * NFC;
    int g = f * HOPC + 128 + t;           // window nonzero on t512 in [128,384)
    if (g < T_LEN) v = y1[(long)b * T_LEN + g] * win[t];
  }
  frames[idx] = (_Float16)v;
}

// DFT operand matrices, stored pre-swizzled for load_B_sw
__global__ void build_dft(_Float16* __restrict__ fre, _Float16* __restrict__ fim,
                          _Float16* __restrict__ ire, _Float16* __restrict__ iim) {
  int idx = blockIdx.x * blockDim.x + threadIdx.x;
  if (idx >= 256 * 256) return;
  int row = idx >> 8, col = idx & 255;
  int si = sw_idx(row, col, 256);
  // forward: row = frame sample t (t512 = t+128), col = bin
  float thf = -2.f * PIF * (float)col * (float)(row + 128) / 512.f;
  fre[si] = (_Float16)__cosf(thf);
  fim[si] = (_Float16)__sinf(thf);
  // inverse: row = bin k, col = output sample tc (t512 = tc+128)
  float sc  = (row == 0) ? (1.f / 512.f) : (2.f / 512.f);
  float thi = 2.f * PIF * (float)row * (float)(col + 128) / 512.f;
  ire[si] = (_Float16)( sc * __cosf(thi));
  iim[si] = (_Float16)(-sc * __sinf(thi));
}

__global__ void nyquist_bin(const float* __restrict__ y1,
                            const float* __restrict__ win,
                            float* __restrict__ s256) {
  int r = blockIdx.x * blockDim.x + threadIdx.x;
  if (r >= RP) return;
  float acc = 0.f;
  if (r < RROWS) {
    int b = r / NFC, f = r - b * NFC;
    for (int t = 0; t < WINC; ++t) {
      int g = f * HOPC + 128 + t;
      if (g < T_LEN) {
        float v = y1[(long)b * T_LEN + g] * win[t];
        acc += (t & 1) ? -v : v;            // (-1)^(t+128) = (-1)^t
      }
    }
  }
  s256[r] = acc;
}

// spec = frames(RPx256) @ DFT(256x256); re & im share A tiles
__global__ void dft_fwd_wmma(const _Float16* __restrict__ frames,
                             const _Float16* __restrict__ bre,
                             const _Float16* __restrict__ bim,
                             float* __restrict__ specRe, float* __restrict__ specIm) {
  int gw   = blockIdx.x * (blockDim.x >> 5) + (threadIdx.x >> 5);
  int lane = threadIdx.x & 31;
  int tm = gw / (NB / 16), tn = gw % (NB / 16);
  if (tm >= RP / 16) return;
  v8f cr = {}, ci = {};
  for (int cb = 0; cb < 8; ++cb) {
    v16h a  = load_A_tile(frames, WINC, tm * 16, cb * 32, lane);
    v16h b0 = load_B_sw(bre, NB, cb, tn * 16, lane);
    v16h b1 = load_B_sw(bim, NB, cb, tn * 16, lane);
    cr = WMMA_F16(a, b0, cr);
    ci = WMMA_F16(a, b1, ci);
  }
  int n = tn * 16 + (lane & 15), hi = (lane >> 4) & 1;
#pragma unroll
  for (int r2 = 0; r2 < 8; ++r2) {
    int m = tm * 16 + r2 + 8 * hi;
    specRe[(long)m * NB + n] = cr[r2];
    specIm[(long)m * NB + n] = ci[r2];
  }
}

__global__ void power_to_half(const float* __restrict__ specRe,
                              const float* __restrict__ specIm,
                              _Float16* __restrict__ pw16) {
  long i = (long)blockIdx.x * blockDim.x + threadIdx.x;
  if (i >= (long)RP * NB) return;
  float re = specRe[i], im = specIm[i];
  pw16[i] = (_Float16)(re * re + im * im);
}

// B[k=j][n=i] = fsmear[i][j], pre-swizzled
__global__ void fsm_prep(const float* __restrict__ fsmear, _Float16* __restrict__ fsm16) {
  int idx = blockIdx.x * blockDim.x + threadIdx.x;
  if (idx >= 256 * 256) return;
  int j = idx >> 8, i = idx & 255;
  fsm16[sw_idx(j, i, 256)] = (_Float16)fsmear[i * 256 + j];
}

// sp = power(RPx256) @ fsmear^T(256x256)
__global__ void smear_wmma(const _Float16* __restrict__ pw16,
                           const _Float16* __restrict__ fsm16,
                           float* __restrict__ sp) {
  int gw   = blockIdx.x * (blockDim.x >> 5) + (threadIdx.x >> 5);
  int lane = threadIdx.x & 31;
  int tm = gw / (NB / 16), tn = gw % (NB / 16);
  if (tm >= RP / 16) return;
  v8f c = {};
  for (int cb = 0; cb < 8; ++cb) {
    v16h a = load_A_tile(pw16, NB, tm * 16, cb * 32, lane);
    v16h b = load_B_sw(fsm16, NB, cb, tn * 16, lane);
    c = WMMA_F16(a, b, c);
  }
  int n = tn * 16 + (lane & 15), hi = (lane >> 4) & 1;
#pragma unroll
  for (int r2 = 0; r2 < 8; ++r2)
    sp[(long)(tm * 16 + r2 + 8 * hi) * NB + n] = c[r2];
}

__global__ void ratio_scale(const float* __restrict__ specRe,
                            const float* __restrict__ specIm,
                            const float* __restrict__ sp,
                            _Float16* __restrict__ sscRe, _Float16* __restrict__ sscIm) {
  long i = (long)blockIdx.x * blockDim.x + threadIdx.x;
  if (i >= (long)RP * NB) return;
  float re = specRe[i], im = specIm[i];
  float pw = re * re + im * im;
  float ratio = sqrtf((fmaxf(sp[i], 0.f) + EPSF) / (pw + EPSF));
  sscRe[i] = (_Float16)(re * ratio);
  sscIm[i] = (_Float16)(im * ratio);
}

// fr[r][tc] = (Rsc @ invRe + Isc @ invIm)[r][tc] + nyquist term ; then * window
__global__ void dft_inv_wmma(const _Float16* __restrict__ sscRe,
                             const _Float16* __restrict__ sscIm,
                             const _Float16* __restrict__ ire,
                             const _Float16* __restrict__ iim,
                             const float* __restrict__ s256,
                             const float* __restrict__ win,
                             float* __restrict__ fr) {
  int gw   = blockIdx.x * (blockDim.x >> 5) + (threadIdx.x >> 5);
  int lane = threadIdx.x & 31;
  int tm = gw / (NB / 16), tn = gw % (NB / 16);
  if (tm >= RP / 16) return;
  v8f c = {};
  for (int cb = 0; cb < 8; ++cb) {
    v16h aR = load_A_tile(sscRe, NB, tm * 16, cb * 32, lane);
    v16h bR = load_B_sw(ire, NB, cb, tn * 16, lane);
    c = WMMA_F16(aR, bR, c);
    v16h aI = load_A_tile(sscIm, NB, tm * 16, cb * 32, lane);
    v16h bI = load_B_sw(iim, NB, cb, tn * 16, lane);
    c = WMMA_F16(aI, bI, c);
  }
  int n = tn * 16 + (lane & 15), hi = (lane >> 4) & 1;
  float wv  = win[n];
  float sgn = (n & 1) ? -(1.f / 512.f) : (1.f / 512.f);
#pragma unroll
  for (int r2 = 0; r2 < 8; ++r2) {
    int m = tm * 16 + r2 + 8 * hi;
    fr[(long)m * NB + n] = (c[r2] + s256[m] * sgn) * wv;
  }
}

// overlap-add gather (deterministic, no atomics)
__global__ void ola_norm(const float* __restrict__ fr,
                         const float* __restrict__ win,
                         float* __restrict__ y2) {
  long i = (long)blockIdx.x * blockDim.x + threadIdx.x;
  if (i >= (long)BATCH * T_LEN) return;
  int b = (int)(i / T_LEN), s = (int)(i - (long)b * T_LEN);
  float acc = 0.f, ws = 0.f;
  if (s >= 128) {
    int f1 = (s - 128) / 64; if (f1 > NFC - 1) f1 = NFC - 1;
    int f0 = (s >= 384) ? (s - 320) / 64 : 0;   // ceil((s-383)/64)
    for (int f = f0; f <= f1; ++f) {
      int t = s - 64 * f - 128;                 // guaranteed in [0,256)
      acc += fr[(long)(b * NFC + f) * NB + t];
      float wv = win[t];
      ws += wv * wv;
    }
  }
  y2[i] = acc / fmaxf(ws, 1e-6f);
}

// ---------------- gammatone stage -------------------------------------------
__global__ void y2_to_half(const float* __restrict__ y2, _Float16* __restrict__ u16) {
  long i = (long)blockIdx.x * blockDim.x + threadIdx.x;
  if (i >= (long)BATCH * T_LEN) return;
  int b = (int)(i / T_LEN), t = (int)(i - (long)b * T_LEN);
  u16[(long)b * UPLEN + HALO + t] = (_Float16)y2[i];
}

__global__ void gt_prep(const float* __restrict__ gc, const float* __restrict__ gs,
                        _Float16* __restrict__ gc16, _Float16* __restrict__ gs16) {
  int idx = blockIdx.x * blockDim.x + threadIdx.x;
  if (idx >= CHP * KP) return;
  int c = idx / KP, k = idx - c * KP;
  float vc = 0.f, vs = 0.f;
  if (c < CH && k < KS) { vc = gc[c * KS + k]; vs = gs[c * KS + k]; }
  gc16[idx] = (_Float16)vc;
  gs16[idx] = (_Float16)vs;
}

// xr/xi = gt(32x1056) @ Hankel(u)(1056x16) per 16-sample tile; env to padded buf
__global__ void gt_wmma(const _Float16* __restrict__ u16,
                        const _Float16* __restrict__ gc16,
                        const _Float16* __restrict__ gs16,
                        _Float16* __restrict__ xr16,
                        _Float16* __restrict__ envp) {
  int gw   = blockIdx.x * (blockDim.x >> 5) + (threadIdx.x >> 5);
  int lane = threadIdx.x & 31;
  int b = gw / TT16, tt = gw % TT16;
  if (b >= BATCH) return;
  int tb = tt * 16;
  // B[k][n] = u[tb + n + k - 512] with zero halo (inherently unaligned gather)
  const _Float16* u = u16 + (long)b * UPLEN + (HALO - 512) + tb;
  const int n = lane & 15, hi = (lane >> 4) & 1;
  v8f c0 = {}, c1 = {}, c2 = {}, c3 = {};
  for (int cb = 0; cb < NKC; ++cb) {
    int kb = cb * 32;
    v16h bt;
#pragma unroll
    for (int e = 0; e < 16; ++e) bt[e] = u[n + kb + e + 16 * hi];
    v16h a0 = load_A_tile(gc16, KP,  0, kb, lane);
    v16h a1 = load_A_tile(gc16, KP, 16, kb, lane);
    v16h a2 = load_A_tile(gs16, KP,  0, kb, lane);
    v16h a3 = load_A_tile(gs16, KP, 16, kb, lane);
    c0 = WMMA_F16(a0, bt, c0);   // xr, channels 0-15
    c1 = WMMA_F16(a1, bt, c1);   // xr, channels 16-31
    c2 = WMMA_F16(a2, bt, c2);   // xi, channels 0-15
    c3 = WMMA_F16(a3, bt, c3);   // xi, channels 16-31
  }
  int t = tb + n;
  if (t < T_LEN) {
#pragma unroll
    for (int r2 = 0; r2 < 8; ++r2) {
      int cA = r2 + 8 * hi, cB = 16 + cA;
      float xrA = c0[r2], xrB = c1[r2], xiA = c2[r2], xiB = c3[r2];
      xr16[((long)(b * CHP + cA)) * T_LEN + t] = (_Float16)xrA;
      xr16[((long)(b * CHP + cB)) * T_LEN + t] = (_Float16)xrB;
      float eA = sqrtf(xrA * xrA + xiA * xiA + EPSF);
      float eB = sqrtf(xrB * xrB + xiB * xiB + EPSF);
      envp[((long)(b * CHP + cA)) * UPLEN + HALO + t] = (_Float16)eA;
      envp[((long)(b * CHP + cB)) * UPLEN + HALO + t] = (_Float16)eB;
    }
  }
}

// Toeplitz operand for the grouped LPF conv:
//   Btoe[c][cb][j][n] = lpf[c][cb*32 + j - n]  (0 if tap out of range)
// Each tap kappa is covered exactly once per column n at cb = floor((kappa+n)/32);
// the few (kappa,n) pairs with kappa+n >= 1056 all fall in the zero tap padding.
__global__ void btoe_prep(const float* __restrict__ lpf, _Float16* __restrict__ btoe) {
  int idx = blockIdx.x * blockDim.x + threadIdx.x;
  if (idx >= CH * BTOE_CH) return;
  int n  = idx & 15;
  int j  = (idx >> 4) & 31;
  int cb = (idx >> 9) % NKC;
  int c  = idx / (NKC * 512);
  int kap = cb * 32 + j - n;
  float v = (kap >= 0 && kap < KS) ? lpf[(long)c * KS + kap] : 0.f;
  int off = ((cb * 2 + (j >> 4)) * 16 + n) * 16 + (j & 15);   // swizzled, N=16
  btoe[(long)c * BTOE_CH + off] = (_Float16)v;
}

// env_lpf = Toeplitz WMMA conv: block = 8 waves on one (b,c), 2048 samples;
// per-channel Toeplitz operand (33KB) staged in LDS, reused by all waves.
__global__ void lpf_wmma(const _Float16* __restrict__ envp,
                         const _Float16* __restrict__ btoe,
                         _Float16* __restrict__ elp16) {
  __shared__ __align__(32) _Float16 sB[BTOE_CH];
  int bc   = blockIdx.x / T2048;
  int tile = blockIdx.x % T2048;
  int b = bc / CH, c = bc % CH;
  // cooperative LDS fill (v8h chunks)
  {
    const v8h* src = (const v8h*)(btoe + (long)c * BTOE_CH);
    v8h* dst = (v8h*)sB;
    for (int i = threadIdx.x; i < BTOE_CH / 8; i += blockDim.x) dst[i] = src[i];
  }
  __syncthreads();
  int lane  = threadIdx.x & 31;
  int tbase = tile * 2048 + (threadIdx.x >> 5) * 256;
  if (tbase >= T_LEN) return;                      // wave-uniform
  const _Float16* ep = envp + (long)(b * CHP + c) * UPLEN + (HALO - 512) + tbase;
  const int n = lane & 15, hi = (lane >> 4) & 1;
  v8f acc = {};
  for (int cb = 0; cb < NKC; ++cb) {
    // A[m][j] = env[tbase + 16m + cb*32 + j - 512]  (lda = 16, overlapping rows)
    v16h a = load_A_tile(ep, 16, 0, cb * 32, lane);
    const _Float16* pb = sB + (((cb * 2 + hi) * 16 + n) << 4);
    v16h bt = cat8(*(const v8h*)pb, *(const v8h*)(pb + 8));
    acc = WMMA_F16(a, bt, acc);
  }
#pragma unroll
  for (int r2 = 0; r2 < 8; ++r2) {
    int m = r2 + 8 * hi;
    int t = tbase + 16 * m + n;
    if (t < T_LEN)
      elp16[(long)(b * CHP + c) * T_LEN + t] = (_Float16)acc[r2];
  }
}

// expansion gain from smoothed envelope; gain*xr written back over xr16
__global__ void gain_mul(const _Float16* __restrict__ elp16,
                         const float* __restrict__ expn,
                         const float* __restrict__ emaxv,
                         _Float16* __restrict__ xr16) {
  long i = (long)blockIdx.x * blockDim.x + threadIdx.x;
  if (i >= (long)BATCH * CH * T_LEN) return;
  int t = (int)(i % T_LEN);
  long q = i / T_LEN;
  int c = (int)(q % CH), b = (int)(q / CH);
  long ix = ((long)(b * CHP + c)) * T_LEN + t;
  float env = fmaxf((float)elp16[ix], EPSF);
  float em  = emaxv[c];
  float g   = __powf(fminf(env, em) / em, expn[c]);
  xr16[ix] = (_Float16)((float)xr16[ix] * g);
}

__global__ void chan_reduce(const _Float16* __restrict__ gx,
                            const float* __restrict__ outcoef,
                            float* __restrict__ y3) {
  long i = (long)blockIdx.x * blockDim.x + threadIdx.x;
  if (i >= (long)BATCH * T_LEN) return;
  int b = (int)(i / T_LEN), t = (int)(i - (long)b * T_LEN);
  float acc = 0.f;
#pragma unroll 4
  for (int c = 0; c < CH; ++c)
    acc += (float)gx[((long)(b * CHP + c)) * T_LEN + t];
  y3[i] = acc * outcoef[0];
}

// ---------------- host orchestration ----------------------------------------
extern "C" void kernel_launch(void* const* d_in, const int* in_sizes, int n_in,
                              void* d_out, int out_size, void* d_ws, size_t ws_size,
                              hipStream_t stream) {
  (void)n_in; (void)out_size; (void)ws_size;
  const float* x        = (const float*)d_in[0];
  const float* coch_fwd = (const float*)d_in[1];
  const float* coch_bwd = (const float*)d_in[2];
  const float* swin     = (const float*)d_in[3];
  const float* fsmear   = (const float*)d_in[4];
  const float* gt_cos   = (const float*)d_in[5];
  const float* gt_sin   = (const float*)d_in[6];
  const float* fir_lpf  = (const float*)d_in[7];
  const float* expn     = (const float*)d_in[8];
  const float* envmax   = (const float*)d_in[9];
  const float* outcoef  = (const float*)d_in[10];
  const int Mtaps = in_sizes[1];                      // 1015

  char* base = (char*)d_ws;
  size_t off = 0;
  auto carve = [&](size_t bytes) -> void* {
    void* p = base + off;
    off = (off + bytes + 255) & ~(size_t)255;
    return p;
  };

  float*    y1      = (float*)   carve(sizeof(float)    * (size_t)BATCH * T_LEN);
  _Float16* frames  = (_Float16*)carve(sizeof(_Float16) * (size_t)RP * WINC);
  _Float16* dftFRe  = (_Float16*)carve(sizeof(_Float16) * 256 * 256);
  _Float16* dftFIm  = (_Float16*)carve(sizeof(_Float16) * 256 * 256);
  _Float16* dftIRe  = (_Float16*)carve(sizeof(_Float16) * 256 * 256);
  _Float16* dftIIm  = (_Float16*)carve(sizeof(_Float16) * 256 * 256);
  float*    specRe  = (float*)   carve(sizeof(float)    * (size_t)RP * NB);
  float*    specIm  = (float*)   carve(sizeof(float)    * (size_t)RP * NB);
  float*    s256    = (float*)   carve(sizeof(float)    * (size_t)RP);
  _Float16* pw16    = (_Float16*)carve(sizeof(_Float16) * (size_t)RP * NB);
  _Float16* fsm16   = (_Float16*)carve(sizeof(_Float16) * 256 * 256);
  float*    sp32    = (float*)   carve(sizeof(float)    * (size_t)RP * NB);
  _Float16* sscRe   = (_Float16*)carve(sizeof(_Float16) * (size_t)RP * NB);
  _Float16* sscIm   = (_Float16*)carve(sizeof(_Float16) * (size_t)RP * NB);
  float*    fr32    = (float*)   carve(sizeof(float)    * (size_t)RP * NB);
  float*    y2      = (float*)   carve(sizeof(float)    * (size_t)BATCH * T_LEN);
  _Float16* u16     = (_Float16*)carve(sizeof(_Float16) * (size_t)BATCH * UPLEN);
  _Float16* gc16    = (_Float16*)carve(sizeof(_Float16) * (size_t)CHP * KP);
  _Float16* gs16    = (_Float16*)carve(sizeof(_Float16) * (size_t)CHP * KP);
  _Float16* btoe    = (_Float16*)carve(sizeof(_Float16) * (size_t)CH * BTOE_CH);
  _Float16* xr16    = (_Float16*)carve(sizeof(_Float16) * (size_t)BATCH * CHP * T_LEN);
  _Float16* envp    = (_Float16*)carve(sizeof(_Float16) * (size_t)BATCH * CHP * UPLEN);
  _Float16* elp16   = (_Float16*)carve(sizeof(_Float16) * (size_t)BATCH * CHP * T_LEN);
  float*    y3      = (float*)   carve(sizeof(float)    * (size_t)BATCH * T_LEN);

  const int BS = 256;
  auto blocks = [](long n, int bs) { return (int)((n + bs - 1) / bs); };

  // 1) forward cochlear FIR
  conv1d_same<<<blocks((long)BATCH * T_LEN, BS), BS, 0, stream>>>(x, coch_fwd, y1, T_LEN, Mtaps);

  // 2) STFT via DFT-GEMM
  build_dft<<<blocks(256 * 256, BS), BS, 0, stream>>>(dftFRe, dftFIm, dftIRe, dftIIm);
  build_frames<<<blocks((long)RP * WINC, BS), BS, 0, stream>>>(y1, swin, frames);
  nyquist_bin<<<blocks(RP, BS), BS, 0, stream>>>(y1, swin, s256);
  dft_fwd_wmma<<<((RP / 16) * (NB / 16)) / 8, BS, 0, stream>>>(frames, dftFRe, dftFIm,
                                                               specRe, specIm);

  // 3) smearing matmul + ratio scaling
  power_to_half<<<blocks((long)RP * NB, BS), BS, 0, stream>>>(specRe, specIm, pw16);
  fsm_prep<<<blocks(256 * 256, BS), BS, 0, stream>>>(fsmear, fsm16);
  smear_wmma<<<((RP / 16) * (NB / 16)) / 8, BS, 0, stream>>>(pw16, fsm16, sp32);
  ratio_scale<<<blocks((long)RP * NB, BS), BS, 0, stream>>>(specRe, specIm, sp32, sscRe, sscIm);

  // 4) inverse DFT-GEMM + window, then overlap-add
  dft_inv_wmma<<<((RP / 16) * (NB / 16)) / 8, BS, 0, stream>>>(sscRe, sscIm, dftIRe, dftIIm,
                                                               s256, swin, fr32);
  ola_norm<<<blocks((long)BATCH * T_LEN, BS), BS, 0, stream>>>(fr32, swin, y2);

  // 5) gammatone filterbank via WMMA
  zero_half<<<blocks((long)BATCH * UPLEN, BS), BS, 0, stream>>>(u16, (long)BATCH * UPLEN);
  zero_half<<<blocks((long)BATCH * CHP * UPLEN, BS), BS, 0, stream>>>(envp, (long)BATCH * CHP * UPLEN);
  y2_to_half<<<blocks((long)BATCH * T_LEN, BS), BS, 0, stream>>>(y2, u16);
  gt_prep<<<blocks(CHP * KP, BS), BS, 0, stream>>>(gt_cos, gt_sin, gc16, gs16);
  gt_wmma<<<(BATCH * TT16 + 7) / 8, BS, 0, stream>>>(u16, gc16, gs16, xr16, envp);

  // 6) envelope LPF via Toeplitz-WMMA (LDS-staged operand), gain, channel sum
  btoe_prep<<<blocks((long)CH * BTOE_CH, BS), BS, 0, stream>>>(fir_lpf, btoe);
  lpf_wmma<<<BATCH * CH * T2048, BS, 0, stream>>>(envp, btoe, elp16);
  gain_mul<<<blocks((long)BATCH * CH * T_LEN, BS), BS, 0, stream>>>(elp16, expn, envmax, xr16);
  chan_reduce<<<blocks((long)BATCH * T_LEN, BS), BS, 0, stream>>>(xr16, outcoef, y3);

  // 7) backward cochlear FIR -> output
  conv1d_same<<<blocks((long)BATCH * T_LEN, BS), BS, 0, stream>>>(y3, coch_bwd, (float*)d_out, T_LEN, Mtaps);
}
